// Faceformer_76982993813511
// MI455X (gfx1250) — compile-verified, matrix operand order checked by reading, not run on hardware
//
#include <hip/hip_runtime.h>
#include <hip/hip_bf16.h>

#define D 1024
#define NH 4
#define HD 256
#define PERIOD 25
#define BS 52
#define NB 32      // decoder persistent blocks
#define BT 256     // decoder threads per block (8 wave32 waves)

typedef __attribute__((ext_vector_type(16))) _Float16 v16h;
typedef __attribute__((ext_vector_type(8)))  float    v8f;

// ---------------------------------------------------------------- utilities

__device__ __forceinline__ float warp_sum(float v) {
  v += __shfl_down(v, 16);
  v += __shfl_down(v, 8);
  v += __shfl_down(v, 4);
  v += __shfl_down(v, 2);
  v += __shfl_down(v, 1);
  return v;
}

__device__ __forceinline__ float blk_sum(float v, float* s_red) {
  int tid = threadIdx.x;
  s_red[tid] = v;
  __syncthreads();
  for (int off = BT / 2; off > 0; off >>= 1) {
    if (tid < off) s_red[tid] += s_red[tid + off];
    __syncthreads();
  }
  float r = s_red[0];
  __syncthreads();
  return r;
}

__device__ __forceinline__ float blk_max(float v, float* s_red) {
  int tid = threadIdx.x;
  s_red[tid] = v;
  __syncthreads();
  for (int off = BT / 2; off > 0; off >>= 1) {
    if (tid < off) s_red[tid] = fmaxf(s_red[tid], s_red[tid + off]);
    __syncthreads();
  }
  float r = s_red[0];
  __syncthreads();
  return r;
}

// Grid-wide barrier: monotone arrival counter; all NB blocks are co-resident.
__device__ __forceinline__ void gbar(int* cnt, int& epoch) {
  __syncthreads();
  __threadfence();                 // release pending global writes
  ++epoch;
  if (threadIdx.x == 0) {
    atomicAdd(cnt, 1);
    while (__hip_atomic_load(cnt, __ATOMIC_RELAXED, __HIP_MEMORY_SCOPE_AGENT) < epoch * NB)
      __builtin_amdgcn_s_sleep(2);
  }
  __syncthreads();
  __threadfence();                 // acquire
}

// ---------------------------------------------------------------- fp32 -> f16

__global__ void k_f32_to_f16(const float* __restrict__ src, _Float16* __restrict__ dst, int n) {
  int i = blockIdx.x * blockDim.x + threadIdx.x;
  int stride = gridDim.x * blockDim.x;
  for (; i < n; i += stride) dst[i] = (_Float16)src[i];
}

// ---------------------------------------------------------------- WMMA GEMM
// C[M x N] = A[M x K] (f16, row-major) * B[N x K] (f16, row-major)^T + bias[N]
// 256 threads = 8 waves arranged 2(M) x 4(N); block tile 32 x 256; wave tile 16 x 64.
// A-lane and B-lane fragments use the same contiguous K-packing (consistent K
// permutation on both operands -> correct dot products).
// K-loop is unrolled x2 with explicit double buffering (fragment sets X/Y) so
// the loads of one set are in flight while the 4 WMMAs of the other execute.
__global__ void __launch_bounds__(256) k_wmma_gemm(
    const _Float16* __restrict__ A, const _Float16* __restrict__ B,
    const float* __restrict__ bias, float* __restrict__ C,
    int M, int N, int K) {
  const int lane  = threadIdx.x & 31;
  const int wave  = threadIdx.x >> 5;
  const int wm    = wave >> 2;                 // 0..1
  const int wn    = wave & 3;                  // 0..3
  const int tileM = blockIdx.y * 32 + wm * 16;
  const int tileN = blockIdx.x * 256 + wn * 64;
  const int r     = lane & 15;
  const int hf    = lane >> 4;

  int arow = tileM + r;
  if (arow >= M) arow = 0;                     // clamp; guarded at store
  const _Float16* Ap  = A + (size_t)arow * K + hf * 16;
  const _Float16* Bp0 = B + (size_t)(tileN +  0 + r) * K + hf * 16;
  const _Float16* Bp1 = B + (size_t)(tileN + 16 + r) * K + hf * 16;
  const _Float16* Bp2 = B + (size_t)(tileN + 32 + r) * K + hf * 16;
  const _Float16* Bp3 = B + (size_t)(tileN + 48 + r) * K + hf * 16;

  v8f acc0 = {}, acc1 = {}, acc2 = {}, acc3 = {};

  // prologue: fragment set X at k = 0
  v16h aX  = *(const v16h*)(Ap);
  v16h b0X = *(const v16h*)(Bp0);
  v16h b1X = *(const v16h*)(Bp1);
  v16h b2X = *(const v16h*)(Bp2);
  v16h b3X = *(const v16h*)(Bp3);

  for (int k = 0; k < K; k += 64) {            // K % 64 == 0
    __builtin_prefetch(Ap  + k + 256, 0, 3);   // WGP-scope, 4 iters ahead
    __builtin_prefetch(Bp0 + k + 256, 0, 3);
    __builtin_prefetch(Bp1 + k + 256, 0, 3);
    __builtin_prefetch(Bp2 + k + 256, 0, 3);
    __builtin_prefetch(Bp3 + k + 256, 0, 3);
    // issue set Y loads (k+32), then consume set X
    v16h aY  = *(const v16h*)(Ap  + k + 32);
    v16h b0Y = *(const v16h*)(Bp0 + k + 32);
    v16h b1Y = *(const v16h*)(Bp1 + k + 32);
    v16h b2Y = *(const v16h*)(Bp2 + k + 32);
    v16h b3Y = *(const v16h*)(Bp3 + k + 32);
    acc0 = __builtin_amdgcn_wmma_f32_16x16x32_f16(false, aX, false, b0X, (short)0, acc0, false, false);
    acc1 = __builtin_amdgcn_wmma_f32_16x16x32_f16(false, aX, false, b1X, (short)0, acc1, false, false);
    acc2 = __builtin_amdgcn_wmma_f32_16x16x32_f16(false, aX, false, b2X, (short)0, acc2, false, false);
    acc3 = __builtin_amdgcn_wmma_f32_16x16x32_f16(false, aX, false, b3X, (short)0, acc3, false, false);
    // issue next set X loads (k+64; redirected in-bounds on last iter), consume set Y
    int k2 = (k + 64 < K) ? (k + 64) : 0;
    aX  = *(const v16h*)(Ap  + k2);
    b0X = *(const v16h*)(Bp0 + k2);
    b1X = *(const v16h*)(Bp1 + k2);
    b2X = *(const v16h*)(Bp2 + k2);
    b3X = *(const v16h*)(Bp3 + k2);
    acc0 = __builtin_amdgcn_wmma_f32_16x16x32_f16(false, aY, false, b0Y, (short)0, acc0, false, false);
    acc1 = __builtin_amdgcn_wmma_f32_16x16x32_f16(false, aY, false, b1Y, (short)0, acc1, false, false);
    acc2 = __builtin_amdgcn_wmma_f32_16x16x32_f16(false, aY, false, b2Y, (short)0, acc2, false, false);
    acc3 = __builtin_amdgcn_wmma_f32_16x16x32_f16(false, aY, false, b3Y, (short)0, acc3, false, false);
  }

  // C/D layout: VGPR v holds row tileM + hf*8 + v, col = tileN + s*16 + (lane%16)
  for (int s = 0; s < 4; ++s) {
    v8f acc = (s == 0) ? acc0 : (s == 1) ? acc1 : (s == 2) ? acc2 : acc3;
    int col = tileN + s * 16 + r;
    float bb = bias[col];
    for (int v = 0; v < 8; ++v) {
      int row = tileM + hf * 8 + v;
      if (row < M) C[(size_t)row * N + col] = acc[v] + bb;
    }
  }
}

// ---------------------------------------------------------------- init: pe, style, x0, sync

__global__ void k_init(const float* __restrict__ one_hot, const float* __restrict__ W_obj,
                       float* __restrict__ pe, float* __restrict__ style,
                       float* __restrict__ x0, int* __restrict__ syncc, int T1) {
  int idx = blockIdx.x * blockDim.x + threadIdx.x;
  int stride = gridDim.x * blockDim.x;
  const float c = -logf(10000.0f) / (float)D;
  for (int i = idx; i < T1 * D; i += stride) {
    int n = i / D, d = i - n * D;
    int k2 = (d >> 1) << 1;                      // 2k
    float ang = (float)(n % PERIOD) * expf((float)k2 * c);
    float v = (d & 1) ? cosf(ang) : sinf(ang);
    pe[i] = v;
    if (n == 0) {
      float st = one_hot[0] * W_obj[d * 2 + 0] + one_hot[1] * W_obj[d * 2 + 1];
      style[d] = st;
      x0[d] = st + v;
    }
  }
  if (idx < 64) syncc[idx] = 0;
}

// ---------------------------------------------------------------- sequential decoder
// Persistent NB x BT kernel; weights (~34 MB fp32) stay L2-resident; 7 grid
// barriers per timestep. Assumes t <= 1024 (harness: t = 600).

__global__ void __launch_bounds__(BT) k_decoder(
    const float* __restrict__ sa_in_w, const float* __restrict__ sa_in_b,
    const float* __restrict__ sa_out_w, const float* __restrict__ sa_out_b,
    const float* __restrict__ ff1_w,  const float* __restrict__ ff1_b,
    const float* __restrict__ ff2_w,  const float* __restrict__ ff2_b,
    const float* __restrict__ W_vmap, const float* __restrict__ b_vmap,
    const float* __restrict__ W_vmap_r, const float* __restrict__ b_vmap_r,
    const float* __restrict__ ln1_g, const float* __restrict__ ln1_b,
    const float* __restrict__ ln2_g, const float* __restrict__ ln2_b,
    const float* __restrict__ ln3_g, const float* __restrict__ ln3_b,
    const float* __restrict__ ca_out, const float* __restrict__ pe,
    const float* __restrict__ style,
    float* __restrict__ xbuf, float* __restrict__ qkv,
    float* __restrict__ Kc, float* __restrict__ Vc,
    float* __restrict__ sa, float* __restrict__ salin,
    float* __restrict__ h2buf, float* __restrict__ ffmid, float* __restrict__ ffout,
    int* __restrict__ syncc, float* __restrict__ out, int t) {
  __shared__ float s_vec[2048];
  __shared__ float s_scores[1024];
  __shared__ float s_red[BT];
  __shared__ float s_bs[64];

  const int tid  = threadIdx.x;
  const int bid  = blockIdx.x;
  const int lane = tid & 31;
  const int warp = tid >> 5;            // 0..7
  const float scale = 0.0625f;          // 1/sqrt(HD)
  int epoch = 0;

  for (int i = 0; i < t; ++i) {
    // ---- Phase A: qkv = x @ sa_in_w^T + b ; fill K/V cache at row i ----
    for (int d = tid; d < D; d += BT) s_vec[d] = xbuf[d];
    __syncthreads();
    for (int m = 0; m < 12; ++m) {                    // 3072 rows / 32 blocks / 8 warps
      int row = bid * 96 + warp + 8 * m;
      const float* wr = sa_in_w + (size_t)row * D;
      float p = 0.f;
      for (int u = 0; u < 32; ++u) { int c = u * 32 + lane; p = fmaf(s_vec[c], wr[c], p); }
      p = warp_sum(p);
      if (lane == 0) {
        float v = p + sa_in_b[row];
        qkv[row] = v;
        if (row >= D && row < 2 * D)      Kc[(size_t)i * D + (row - D)]     = v;
        else if (row >= 2 * D)            Vc[(size_t)i * D + (row - 2 * D)] = v;
      }
    }
    gbar(syncc, epoch);

    // ---- Phase B: self-attention, one head per block (blocks 0..3) ----
    if (bid < NH) {
      const int h = bid;
      const int n = i + 1;
      const float slope = exp2f(-2.0f * (float)(h + 1));   // 0.25, 0.0625, ...
      for (int d = tid; d < HD; d += BT) s_vec[d] = qkv[h * HD + d];
      __syncthreads();
      for (int j = warp; j < n; j += 8) {
        const float* kr = Kc + (size_t)j * D + h * HD;
        float p = 0.f;
        for (int u = 0; u < 8; ++u) { int c = u * 32 + lane; p = fmaf(s_vec[c], kr[c], p); }
        p = warp_sum(p);
        if (lane == 0)
          s_scores[j] = p * scale + slope * (float)(-((i - j) / PERIOD));
      }
      __syncthreads();
      float lm = -3.0e38f;
      for (int j = tid; j < n; j += BT) lm = fmaxf(lm, s_scores[j]);
      float mx = blk_max(lm, s_red);
      float ls = 0.f;
      for (int j = tid; j < n; j += BT) { float e = expf(s_scores[j] - mx); s_scores[j] = e; ls += e; }
      float inv = 1.0f / blk_sum(ls, s_red);
      {
        float acc = 0.f;
        int d = tid;                                  // HD == BT
        for (int j = 0; j < n; ++j)
          acc = fmaf(s_scores[j], Vc[(size_t)j * D + h * HD + d], acc);
        sa[h * HD + d] = acc * inv;
      }
    }
    gbar(syncc, epoch);

    // ---- Phase C: salin = sa @ sa_out_w^T + b ----
    for (int d = tid; d < D; d += BT) s_vec[d] = sa[d];
    __syncthreads();
    for (int m = 0; m < 4; ++m) {
      int row = bid * 32 + warp + 8 * m;
      const float* wr = sa_out_w + (size_t)row * D;
      float p = 0.f;
      for (int u = 0; u < 32; ++u) { int c = u * 32 + lane; p = fmaf(s_vec[c], wr[c], p); }
      p = warp_sum(p);
      if (lane == 0) salin[row] = p + sa_out_b[row];
    }
    gbar(syncc, epoch);

    // ---- Phase D (block 0): h1 = LN1(x + salin); h2 = LN2(h1 + ca_out[i]) ----
    if (bid == 0) {
      float lsum = 0.f, lsq = 0.f;
      for (int d = tid; d < D; d += BT) {
        float y = xbuf[d] + salin[d];
        s_vec[d] = y; lsum += y; lsq += y * y;
      }
      float ssm = blk_sum(lsum, s_red);
      float ssq = blk_sum(lsq, s_red);
      float mu = ssm / (float)D;
      float rs = rsqrtf(ssq / (float)D - mu * mu + 1e-5f);
      float l2s = 0.f, l2q = 0.f;
      for (int d = tid; d < D; d += BT) {
        float h1 = (s_vec[d] - mu) * rs * ln1_g[d] + ln1_b[d];
        float z  = h1 + ca_out[(size_t)i * D + d];
        s_vec[d] = z; l2s += z; l2q += z * z;
      }
      float s2 = blk_sum(l2s, s_red);
      float q2 = blk_sum(l2q, s_red);
      float mu2 = s2 / (float)D;
      float rs2 = rsqrtf(q2 / (float)D - mu2 * mu2 + 1e-5f);
      for (int d = tid; d < D; d += BT)
        h2buf[d] = (s_vec[d] - mu2) * rs2 * ln2_g[d] + ln2_b[d];
    }
    gbar(syncc, epoch);

    // ---- Phase E: ffmid = relu(h2 @ ff1_w^T + b) ----
    for (int d = tid; d < D; d += BT) s_vec[d] = h2buf[d];
    __syncthreads();
    for (int m = 0; m < 8; ++m) {                     // 2048 rows
      int row = bid * 64 + warp + 8 * m;
      const float* wr = ff1_w + (size_t)row * D;
      float p = 0.f;
      for (int u = 0; u < 32; ++u) { int c = u * 32 + lane; p = fmaf(s_vec[c], wr[c], p); }
      p = warp_sum(p);
      if (lane == 0) { float v = p + ff1_b[row]; ffmid[row] = v > 0.f ? v : 0.f; }
    }
    gbar(syncc, epoch);

    // ---- Phase F: ffout = ffmid @ ff2_w^T + b ----
    for (int d = tid; d < 2 * D; d += BT) s_vec[d] = ffmid[d];
    __syncthreads();
    for (int m = 0; m < 4; ++m) {
      int row = bid * 32 + warp + 8 * m;
      const float* wr = ff2_w + (size_t)row * 2 * D;
      float p = 0.f;
      for (int u = 0; u < 64; ++u) { int c = u * 32 + lane; p = fmaf(s_vec[c], wr[c], p); }
      p = warp_sum(p);
      if (lane == 0) ffout[row] = p + ff2_b[row];
    }
    gbar(syncc, epoch);

    // ---- Phase G (block 0): h3 = LN3(h2 + ffout); bs_i; x_next ----
    if (bid == 0) {
      float lsum = 0.f, lsq = 0.f;
      for (int d = tid; d < D; d += BT) {
        float y = h2buf[d] + ffout[d];
        s_vec[d] = y; lsum += y; lsq += y * y;
      }
      float ssm = blk_sum(lsum, s_red);
      float ssq = blk_sum(lsq, s_red);
      float mu = ssm / (float)D;
      float rs = rsqrtf(ssq / (float)D - mu * mu + 1e-5f);
      for (int d = tid; d < D; d += BT)
        s_vec[d] = (s_vec[d] - mu) * rs * ln3_g[d] + ln3_b[d];   // h3
      __syncthreads();
      for (int row = warp; row < BS; row += 8) {
        const float* wr = W_vmap_r + (size_t)row * D;
        float p = 0.f;
        for (int u = 0; u < 32; ++u) { int c = u * 32 + lane; p = fmaf(s_vec[c], wr[c], p); }
        p = warp_sum(p);
        if (lane == 0) {
          float v = p + b_vmap_r[row];
          s_bs[row] = v;
          out[(size_t)i * BS + row] = v;
        }
      }
      __syncthreads();
      for (int d = tid; d < D; d += BT) {
        float acc = b_vmap[d] + style[d] + pe[(size_t)(i + 1) * D + d];
        const float* wv = W_vmap + (size_t)d * BS;
        for (int s = 0; s < BS; ++s) acc = fmaf(s_bs[s], wv[s], acc);
        xbuf[d] = acc;
      }
    }
    gbar(syncc, epoch);
  }
}

// ---------------------------------------------------------------- launch

extern "C" void kernel_launch(void* const* d_in, const int* in_sizes, int n_in,
                              void* d_out, int out_size, void* d_ws, size_t ws_size,
                              hipStream_t stream) {
  const float* one_hot  = (const float*)d_in[0];
  const float* audio    = (const float*)d_in[1];
  const float* W_audio  = (const float*)d_in[2];
  const float* b_audio  = (const float*)d_in[3];
  const float* W_obj    = (const float*)d_in[4];
  const float* W_vmap   = (const float*)d_in[5];
  const float* b_vmap   = (const float*)d_in[6];
  const float* W_vmap_r = (const float*)d_in[7];
  const float* b_vmap_r = (const float*)d_in[8];
  const float* sa_in_w  = (const float*)d_in[9];
  const float* sa_in_b  = (const float*)d_in[10];
  const float* sa_out_w = (const float*)d_in[11];
  const float* sa_out_b = (const float*)d_in[12];
  const float* ca_in_w  = (const float*)d_in[13];
  const float* ca_in_b  = (const float*)d_in[14];
  const float* ca_out_w = (const float*)d_in[15];
  const float* ca_out_b = (const float*)d_in[16];
  const float* ff1_w    = (const float*)d_in[17];
  const float* ff1_b    = (const float*)d_in[18];
  const float* ff2_w    = (const float*)d_in[19];
  const float* ff2_b    = (const float*)d_in[20];
  const float* ln1_g    = (const float*)d_in[21];
  const float* ln1_b    = (const float*)d_in[22];
  const float* ln2_g    = (const float*)d_in[23];
  const float* ln2_b    = (const float*)d_in[24];
  const float* ln3_g    = (const float*)d_in[25];
  const float* ln3_b    = (const float*)d_in[26];
  float* out = (float*)d_out;
  const int T = in_sizes[1] / 768;
  (void)n_in; (void)out_size; (void)ws_size;

  char* p = (char*)d_ws;
  auto carve = [&](size_t bytes) { char* r = p; p += (bytes + 255) & ~(size_t)255; return r; };
  _Float16* a16    = (_Float16*)carve((size_t)T * 768 * 2);
  _Float16* wa16   = (_Float16*)carve((size_t)1024 * 768 * 2);
  float*    hs32   = (float*)   carve((size_t)T * D * 4);
  _Float16* hs16   = (_Float16*)carve((size_t)T * D * 2);
  _Float16* wvca16 = (_Float16*)carve((size_t)D * D * 2);
  float*    catmp  = (float*)   carve((size_t)T * D * 4);
  _Float16* cat16  = (_Float16*)carve((size_t)T * D * 2);
  _Float16* wcao16 = (_Float16*)carve((size_t)D * D * 2);
  float*    caout  = (float*)   carve((size_t)T * D * 4);
  float*    pe     = (float*)   carve((size_t)(T + 1) * D * 4);
  float*    style  = (float*)   carve((size_t)D * 4);
  float*    xbuf   = (float*)   carve((size_t)D * 4);
  float*    qkv    = (float*)   carve((size_t)3 * D * 4);
  float*    Kc     = (float*)   carve((size_t)T * D * 4);
  float*    Vc     = (float*)   carve((size_t)T * D * 4);
  float*    sa     = (float*)   carve((size_t)D * 4);
  float*    salin  = (float*)   carve((size_t)D * 4);
  float*    h2b    = (float*)   carve((size_t)D * 4);
  float*    ffm    = (float*)   carve((size_t)2 * D * 4);
  float*    ffo    = (float*)   carve((size_t)D * 4);
  int*      syncc  = (int*)     carve(256);

  // f16 staging for WMMA GEMMs
  k_f32_to_f16<<<256, 256, 0, stream>>>(audio, a16, T * 768);
  k_f32_to_f16<<<256, 256, 0, stream>>>(W_audio, wa16, 1024 * 768);
  k_f32_to_f16<<<256, 256, 0, stream>>>(ca_in_w + (size_t)2 * D * D, wvca16, D * D);  // Wv_ca
  k_f32_to_f16<<<256, 256, 0, stream>>>(ca_out_w, wcao16, D * D);

  dim3 g1(D / 256, (T + 31) / 32);
  // hs = audio @ W_audio^T + b_audio
  k_wmma_gemm<<<g1, 256, 0, stream>>>(a16, wa16, b_audio, hs32, T, D, 768);
  k_f32_to_f16<<<256, 256, 0, stream>>>(hs32, hs16, T * D);
  // catmp = hs @ Wv_ca^T + bv_ca
  k_wmma_gemm<<<g1, 256, 0, stream>>>(hs16, wvca16, ca_in_b + 2 * D, catmp, T, D, D);
  k_f32_to_f16<<<256, 256, 0, stream>>>(catmp, cat16, T * D);
  // ca_out = catmp @ ca_out_w^T + ca_out_b
  k_wmma_gemm<<<g1, 256, 0, stream>>>(cat16, wcao16, ca_out_b, caout, T, D, D);

  // pe / style / x0 / barrier reset (re-done every call: graph-replay safe)
  k_init<<<256, 256, 0, stream>>>(one_hot, W_obj, pe, style, xbuf, syncc, T + 1);

  // sequential decoder: persistent cooperative kernel
  k_decoder<<<NB, BT, 0, stream>>>(sa_in_w, sa_in_b, sa_out_w, sa_out_b,
                                   ff1_w, ff1_b, ff2_w, ff2_b,
                                   W_vmap, b_vmap, W_vmap_r, b_vmap_r,
                                   ln1_g, ln1_b, ln2_g, ln2_b, ln3_g, ln3_b,
                                   caout, pe, style,
                                   xbuf, qkv, Kc, Vc, sa, salin, h2b, ffm, ffo,
                                   syncc, out, T);
}
